// Attention_agent_df_50525995270753
// MI455X (gfx1250) — matrix operand order repaired and checked by reading, not compiled, not run on hardware
//
#include <hip/hip_runtime.h>
#include <hip/hip_bf16.h>

typedef __bf16 bf16_t;
typedef __bf16 v16bf __attribute__((ext_vector_type(16)));
typedef __bf16 v8bf  __attribute__((ext_vector_type(8)));
typedef float  v8f   __attribute__((ext_vector_type(8)));

#define THETA 0.7f
#define SCALE 0.17677669529663689f  // 32^-0.5

// Padded image geometry: [8][66][66][256] bf16, origin at (+1,+1).
#define PADW 66
#define PAD_IMG_ELEMS ((size_t)PADW * PADW * 256)

// ---------------- WMMA helpers (CDNA5 wave32, 16x16x32 bf16 -> f32) ----------

__device__ __forceinline__ v8f zero8() {
  v8f z;
#pragma unroll
  for (int i = 0; i < 8; ++i) z[i] = 0.0f;
  return z;
}

// A fragment: source [M][K] row-major, contiguous K. lane L: row = L&15,
// K chunks {kb..kb+7} and {kb+16..kb+23}, kb = (L>=16 ? 8 : 0).
__device__ __forceinline__ v16bf frag_load_A(const bf16_t* base, int rowStride, int lane) {
  int m  = lane & 15;
  int kb = (lane >> 4) << 3;
  const bf16_t* p = base + (size_t)m * rowStride + kb;
  union { v16bf v; v8bf h[2]; } u;
  u.h[0] = *reinterpret_cast<const v8bf*>(p);
  u.h[1] = *reinterpret_cast<const v8bf*>(p + 16);
  return u.v;
}

// B fragment: element (k, n) = srcT[n][k] with srcT [N][K] contiguous-K.
// lane L: col = L&15, K chunks {kb..kb+7} and {kb+16..kb+23}, kb = (L>=16 ? 8 : 0).
__device__ __forceinline__ v16bf frag_load_B(const bf16_t* base, int rowStride, int lane) {
  int n  = lane & 15;
  int kb = (lane >> 4) << 3;
  const bf16_t* p = base + (size_t)n * rowStride + kb;
  union { v16bf v; v8bf h[2]; } u;
  u.h[0] = *reinterpret_cast<const v8bf*>(p);
  u.h[1] = *reinterpret_cast<const v8bf*>(p + 16);
  return u.v;
}

__device__ __forceinline__ v8f wmma_bf16(v16bf a, v16bf b, v8f c) {
  return __builtin_amdgcn_wmma_f32_16x16x32_bf16(false, a, false, b, (short)0, c,
                                                 false, false);
}

// ---------------- workspace zero-fill (16B granules) ----------------------

__global__ void k_zero16(float4* __restrict__ dst, unsigned n16) {
  unsigned i = blockIdx.x * 256 + threadIdx.x;
  if (i < n16) {
    float4 z = {0.0f, 0.0f, 0.0f, 0.0f};
    dst[i] = z;
  }
}

// ---------------- weight prep --------------------------------------------

// Fold theta*hv-sum into center tap, fold BN scale, convert to bf16, tap-major.
__global__ void k_prep_qkv_w(const float* __restrict__ qkv_w,
                             const float* __restrict__ qkv_s,
                             bf16_t* __restrict__ Wq) {
  int idx = blockIdx.x * 256 + threadIdx.x;  // o*256 + i, o < 768
  if (idx >= 768 * 256) return;
  int o = idx >> 8, i = idx & 255;
  const float* w = qkv_w + (size_t)idx * 9;
  float s = qkv_s[o];
  float sum5 = w[1] + w[3] + w[4] + w[5] + w[7];
#pragma unroll
  for (int t = 0; t < 9; ++t) {
    float wt = w[t];
    if (t == 4) wt -= THETA * sum5;
    Wq[((size_t)t * 768 + o) * 256 + i] = (bf16_t)(wt * s);
  }
}

__global__ void k_prep_proj_w(const float* __restrict__ proj_w,
                              const float* __restrict__ proj_s,
                              bf16_t* __restrict__ Wp) {
  int idx = blockIdx.x * 256 + threadIdx.x;  // o*256 + i
  if (idx >= 256 * 256) return;
  Wp[idx] = (bf16_t)(proj_w[idx] * proj_s[idx >> 8]);
}

// NCHW f32 [8][256][4096] -> padded NHWC bf16 [8][66][66][256] (halo untouched).
__global__ void k_x2padded(const float* __restrict__ src, bf16_t* __restrict__ dst) {
  int bid = blockIdx.x;          // 8 * 8 * 64 = 4096 blocks
  int b = bid >> 9;
  int r = bid & 511;
  int c0 = (r >> 6) * 32;
  int y  = r & 63;               // one image row per block
  __shared__ float t[32][65];
  int tid = threadIdx.x;
#pragma unroll
  for (int i = 0; i < 8; ++i) {
    int idx = tid + i * 256;
    int cc = idx >> 6, xx = idx & 63;
    t[cc][xx] = src[((size_t)(b * 256 + c0 + cc)) * 4096 + y * 64 + xx];
  }
  __syncthreads();
#pragma unroll
  for (int i = 0; i < 8; ++i) {
    int idx = tid + i * 256;
    int xx = idx >> 5, cc = idx & 31;
    dst[(((size_t)b * PADW + y + 1) * PADW + xx + 1) * 256 + c0 + cc] = (bf16_t)t[cc][xx];
  }
}

// NCHW f32 [8][256][4096] -> NHWC bf16 [8][4096][256] (unpadded, for proj input).
__global__ void k_nchw2nhwc_bf16(const float* __restrict__ src, bf16_t* __restrict__ dst) {
  int bid = blockIdx.x;          // 4096 blocks
  int b = bid >> 9;
  int r = bid & 511;
  int c0 = (r >> 6) * 32;
  int N0 = (r & 63) * 64;
  __shared__ float t[32][65];
  int tid = threadIdx.x;
#pragma unroll
  for (int i = 0; i < 8; ++i) {
    int idx = tid + i * 256;
    int cc = idx >> 6, nn = idx & 63;
    t[cc][nn] = src[((size_t)(b * 256 + c0 + cc)) * 4096 + N0 + nn];
  }
  __syncthreads();
#pragma unroll
  for (int i = 0; i < 8; ++i) {
    int idx = tid + i * 256;
    int nn = idx >> 5, cc = idx & 31;
    dst[((size_t)(b * 4096 + N0 + nn)) * 256 + c0 + cc] = (bf16_t)t[cc][nn];
  }
}

// ---------------- qkv conv: implicit GEMM, WMMA bf16 ----------------------
// One wave computes a 32(oc) x 64(pixels = one image row) tile. Zero-halo
// image => every fragment load is unconditional and uniform (no EXEC dance).

__global__ void __launch_bounds__(256) k_qkv_conv(const bf16_t* __restrict__ xbP,
                                                  const bf16_t* __restrict__ Wq,
                                                  const float* __restrict__ qkv_b,
                                                  float* __restrict__ q,
                                                  bf16_t* __restrict__ kT,
                                                  float* __restrict__ v) {
  int wave = (blockIdx.x * 256 + threadIdx.x) >> 5;  // 12288 waves
  int lane = threadIdx.x & 31;
  int b   = wave / (24 * 64);
  int r   = wave % (24 * 64);
  int oc0 = (r >> 6) * 32;   // 0..736, step 32
  int y   = r & 63;
  const bf16_t* img = xbP + (size_t)b * PAD_IMG_ELEMS;
  v8f acc[8];
#pragma unroll
  for (int i = 0; i < 8; ++i) acc[i] = zero8();

#pragma unroll
  for (int t = 0; t < 9; ++t) {
    const int dy = t / 3 - 1, dx = t % 3 - 1;
    const bf16_t* rowB  = img + ((size_t)(y + dy + 1) * PADW + (1 + dx)) * 256;
    const bf16_t* wBase = Wq + ((size_t)t * 768 + oc0) * 256;
    if (t < 8) __builtin_prefetch(wBase + (size_t)768 * 256, 0, 1);
#pragma unroll 2
    for (int ic0 = 0; ic0 < 256; ic0 += 32) {
      v16bf a0  = frag_load_A(wBase + ic0, 256, lane);
      v16bf a1  = frag_load_A(wBase + 16 * 256 + ic0, 256, lane);
      v16bf bb0 = frag_load_B(rowB + ic0, 256, lane);
      v16bf bb1 = frag_load_B(rowB + (size_t)16 * 256 + ic0, 256, lane);
      v16bf bb2 = frag_load_B(rowB + (size_t)32 * 256 + ic0, 256, lane);
      v16bf bb3 = frag_load_B(rowB + (size_t)48 * 256 + ic0, 256, lane);
      acc[0] = wmma_bf16(a0, bb0, acc[0]);
      acc[1] = wmma_bf16(a1, bb0, acc[1]);
      acc[2] = wmma_bf16(a0, bb1, acc[2]);
      acc[3] = wmma_bf16(a1, bb1, acc[3]);
      acc[4] = wmma_bf16(a0, bb2, acc[4]);
      acc[5] = wmma_bf16(a1, bb2, acc[5]);
      acc[6] = wmma_bf16(a0, bb3, acc[6]);
      acc[7] = wmma_bf16(a1, bb3, acc[7]);
    }
  }

  int col = lane & 15;
  int mb  = (lane >> 4) << 3;
#pragma unroll
  for (int g = 0; g < 2; ++g) {
#pragma unroll
    for (int j = 0; j < 4; ++j) {
      v8f c = acc[j * 2 + g];
#pragma unroll
      for (int vv = 0; vv < 8; ++vv) {
        int oc = oc0 + g * 16 + mb + vv;
        float val = c[vv] + qkv_b[oc];
        int pix = y * 64 + j * 16 + col;
        if (oc0 < 256) {
          q[((size_t)b * 256 + oc) * 4096 + pix] = val;
        } else if (oc0 < 512) {
          int cc = oc - 256;
          kT[(((size_t)b * 8 + (cc >> 5)) * 4096 + pix) * 32 + (cc & 31)] = (bf16_t)val;
        } else {
          v[((size_t)b * 256 + (oc - 512)) * 4096 + pix] = val;
        }
      }
    }
  }
}

// ---------------- agent pooling -------------------------------------------

__global__ void k_agent_pool(const float* __restrict__ q, float* __restrict__ aPool,
                             bf16_t* __restrict__ aT) {
  int idx = blockIdx.x * 256 + threadIdx.x;  // 8*256*64
  if (idx >= 8 * 256 * 64) return;
  int nn = idx & 63;
  int c  = (idx >> 6) & 255;
  int b  = idx >> 14;
  int py = nn >> 3, px = nn & 7;
  const float* qb = q + ((size_t)b * 256 + c) * 4096;
  float s = 0.0f;
#pragma unroll
  for (int iy = 0; iy < 8; ++iy)
#pragma unroll
    for (int ix = 0; ix < 8; ++ix)
      s += qb[(py * 8 + iy) * 64 + (px * 8 + ix)];
  s *= (1.0f / 64.0f);
  aPool[((size_t)b * 256 + c) * 64 + nn] = s;
  int h = c >> 5, d = c & 31;
  aT[(((size_t)b * 8 + h) * 64 + nn) * 32 + d] = (bf16_t)s;
}

// ---------------- stage 1 logits: aT @ k (K = d = 32) ---------------------
// One wave: all 64 agents (4 A frags, loaded once) x 32 pixels (2 B frags).

__global__ void __launch_bounds__(256) k_stage1(const bf16_t* __restrict__ aT,
                                                const bf16_t* __restrict__ kT,
                                                float* __restrict__ attnK) {
  int wave = (blockIdx.x * 256 + threadIdx.x) >> 5;  // 8192 waves
  int lane = threadIdx.x & 31;
  int bh = wave >> 7;
  int p0 = (wave & 127) * 32;
  const bf16_t* aB = aT + (size_t)bh * 64 * 32;
  v16bf a[4];
#pragma unroll
  for (int i = 0; i < 4; ++i) a[i] = frag_load_A(aB + (size_t)i * 16 * 32, 32, lane);
  v8f acc[8];
#pragma unroll
  for (int i = 0; i < 8; ++i) acc[i] = zero8();
#pragma unroll
  for (int j = 0; j < 2; ++j) {
    v16bf bb = frag_load_B(kT + ((size_t)bh * 4096 + p0 + j * 16) * 32, 32, lane);
#pragma unroll
    for (int i = 0; i < 4; ++i) acc[j * 4 + i] = wmma_bf16(a[i], bb, acc[j * 4 + i]);
  }
  int col = lane & 15;
  int mb  = (lane >> 4) << 3;
#pragma unroll
  for (int j = 0; j < 2; ++j)
#pragma unroll
    for (int i = 0; i < 4; ++i)
#pragma unroll
      for (int vv = 0; vv < 8; ++vv)
        attnK[((size_t)bh * 64 + i * 16 + mb + vv) * 4096 + p0 + j * 16 + col] =
            acc[j * 4 + i][vv] * SCALE;
}

// ---------------- row softmax over 4096 -----------------------------------

__global__ void __launch_bounds__(256) k_softmax(float* __restrict__ attnK) {
  float* p = attnK + (size_t)blockIdx.x * 4096;  // 4096 rows
  int tid = threadIdx.x;
  __shared__ float red[256];
  float m = -1e30f;
  for (int i = tid; i < 4096; i += 256) m = fmaxf(m, p[i]);
  red[tid] = m;
  __syncthreads();
  for (int o = 128; o > 0; o >>= 1) {
    if (tid < o) red[tid] = fmaxf(red[tid], red[tid + o]);
    __syncthreads();
  }
  float M = red[0];
  __syncthreads();
  float s = 0.0f;
  for (int i = tid; i < 4096; i += 256) s += __expf(p[i] - M);
  red[tid] = s;
  __syncthreads();
  for (int o = 128; o > 0; o >>= 1) {
    if (tid < o) red[tid] += red[tid + o];
    __syncthreads();
  }
  float inv = 1.0f / red[0];
  for (int i = tid; i < 4096; i += 256) p[i] = __expf(p[i] - M) * inv;
}

// ---------------- attn aggregation: attn[nn][d] = sum_N attnK * v ---------

__global__ void __launch_bounds__(256) k_agg(const float* __restrict__ attnK,
                                             const float* __restrict__ v,
                                             float* __restrict__ attn) {
  int row = blockIdx.x;  // (b*8+h)*64+nn, 4096 rows
  int bh = row >> 6;
  int b = bh >> 3, h = bh & 7;
  int tid = threadIdx.x;
  const float* pK = attnK + (size_t)row * 4096;
  const float* vB = v + ((size_t)b * 256 + h * 32) * 4096;
  float acc[32];
#pragma unroll
  for (int d = 0; d < 32; ++d) acc[d] = 0.0f;
  for (int N = tid; N < 4096; N += 256) {
    float w = pK[N];
#pragma unroll
    for (int d = 0; d < 32; ++d) acc[d] += w * vB[(size_t)d * 4096 + N];
  }
  __shared__ float red[256][32];
#pragma unroll
  for (int d = 0; d < 32; ++d) red[tid][d] = acc[d];
  __syncthreads();
  for (int o = 128; o > 0; o >>= 1) {
    if (tid < o)
#pragma unroll
      for (int d = 0; d < 32; ++d) red[tid][d] += red[tid + o][d];
    __syncthreads();
  }
  if (tid < 32) attn[(size_t)row * 32 + tid] = red[0][tid];
}

// ---------------- stage 2 fused: logits+softmax+output --------------------

__global__ void __launch_bounds__(256) k_stage2(const float* __restrict__ q,
                                                const float* __restrict__ aPool,
                                                const float* __restrict__ attn,
                                                float* __restrict__ attnOut) {
  int bid = blockIdx.x;  // 8*8*16 = 1024 blocks
  int b = bid >> 7;
  int h = (bid >> 4) & 7;
  int chunk = bid & 15;
  int tid = threadIdx.x;
  __shared__ float aL[64][32];   // [nn][d]
  __shared__ float atL[64][32];  // [nn][d]
  const float* aP = aPool + ((size_t)b * 256 + h * 32) * 64;  // [d][nn]
  const float* at = attn + ((size_t)(b * 8 + h) * 64) * 32;   // [nn][d]
#pragma unroll
  for (int i = 0; i < 8; ++i) {
    int idx = tid + i * 256;
    int nn = idx >> 5, d = idx & 31;
    aL[nn][d]  = aP[(size_t)d * 64 + nn];
    atL[nn][d] = at[idx];
  }
  __syncthreads();
  int N = chunk * 256 + tid;
  const float* qB = q + ((size_t)b * 256 + h * 32) * 4096;
  float qv[32];
#pragma unroll
  for (int d = 0; d < 32; ++d) qv[d] = qB[(size_t)d * 4096 + N];
  float l[64];
  float M = -1e30f;
#pragma unroll
  for (int nn = 0; nn < 64; ++nn) {
    float s = 0.0f;
#pragma unroll
    for (int d = 0; d < 32; ++d) s += qv[d] * aL[nn][d];
    s *= SCALE;
    l[nn] = s;
    M = fmaxf(M, s);
  }
  float sum = 0.0f;
#pragma unroll
  for (int nn = 0; nn < 64; ++nn) { l[nn] = __expf(l[nn] - M); sum += l[nn]; }
  float inv = 1.0f / sum;
  float out[32];
#pragma unroll
  for (int d = 0; d < 32; ++d) out[d] = 0.0f;
#pragma unroll
  for (int nn = 0; nn < 64; ++nn) {
    float w = l[nn] * inv;
#pragma unroll
    for (int d = 0; d < 32; ++d) out[d] += w * atL[nn][d];
  }
  float* oB = attnOut + ((size_t)b * 256 + h * 32) * 4096;
#pragma unroll
  for (int d = 0; d < 32; ++d) oB[(size_t)d * 4096 + N] = out[d];
}

// ---------------- depthwise pe conv + add ---------------------------------

__global__ void __launch_bounds__(256) k_pe_add(const float* __restrict__ v,
                                                const float* __restrict__ attnOut,
                                                const float* __restrict__ pe_w,
                                                const float* __restrict__ pe_s,
                                                const float* __restrict__ pe_b,
                                                float* __restrict__ yf) {
  int bid = blockIdx.x;  // 8*256*16 = 32768
  int b = bid >> 12;
  int c = (bid >> 4) & 255;
  int chunk = bid & 15;
  int N = chunk * 256 + threadIdx.x;
  int y = N >> 6, x = N & 63;
  const float* vB = v + ((size_t)b * 256 + c) * 4096;
  const float* w = pe_w + c * 9;
  float s = 0.0f;
#pragma unroll
  for (int t = 0; t < 9; ++t) {
    int dy = t / 3 - 1, dx = t % 3 - 1;
    int yy = y + dy, xx = x + dx;
    if (yy >= 0 && yy < 64 && xx >= 0 && xx < 64) s += w[t] * vB[yy * 64 + xx];
  }
  size_t o = ((size_t)b * 256 + c) * 4096 + N;
  yf[o] = attnOut[o] + s * pe_s[c] + pe_b[c];
}

// ---------------- 1x1 projection: WMMA GEMM (32oc x 64px per wave) --------

__global__ void __launch_bounds__(256) k_proj(const bf16_t* __restrict__ yb,
                                              const bf16_t* __restrict__ Wp,
                                              const float* __restrict__ proj_b,
                                              float* __restrict__ out) {
  int wave = (blockIdx.x * 256 + threadIdx.x) >> 5;  // 4096 waves
  int lane = threadIdx.x & 31;
  int b   = wave >> 9;
  int r   = wave & 511;
  int oc0 = (r >> 6) * 32;
  int y   = r & 63;
  const bf16_t* yB = yb + ((size_t)b * 4096 + y * 64) * 256;
  v8f acc[8];
#pragma unroll
  for (int i = 0; i < 8; ++i) acc[i] = zero8();
#pragma unroll 2
  for (int ic0 = 0; ic0 < 256; ic0 += 32) {
    v16bf a0  = frag_load_A(Wp + (size_t)oc0 * 256 + ic0, 256, lane);
    v16bf a1  = frag_load_A(Wp + (size_t)(oc0 + 16) * 256 + ic0, 256, lane);
    v16bf bb0 = frag_load_B(yB + ic0, 256, lane);
    v16bf bb1 = frag_load_B(yB + (size_t)16 * 256 + ic0, 256, lane);
    v16bf bb2 = frag_load_B(yB + (size_t)32 * 256 + ic0, 256, lane);
    v16bf bb3 = frag_load_B(yB + (size_t)48 * 256 + ic0, 256, lane);
    acc[0] = wmma_bf16(a0, bb0, acc[0]);
    acc[1] = wmma_bf16(a1, bb0, acc[1]);
    acc[2] = wmma_bf16(a0, bb1, acc[2]);
    acc[3] = wmma_bf16(a1, bb1, acc[3]);
    acc[4] = wmma_bf16(a0, bb2, acc[4]);
    acc[5] = wmma_bf16(a1, bb2, acc[5]);
    acc[6] = wmma_bf16(a0, bb3, acc[6]);
    acc[7] = wmma_bf16(a1, bb3, acc[7]);
  }
  int col = lane & 15;
  int mb  = (lane >> 4) << 3;
#pragma unroll
  for (int g = 0; g < 2; ++g)
#pragma unroll
    for (int j = 0; j < 4; ++j) {
      v8f c = acc[j * 2 + g];
#pragma unroll
      for (int vv = 0; vv < 8; ++vv) {
        int oc = oc0 + g * 16 + mb + vv;
        out[((size_t)b * 256 + oc) * 4096 + y * 64 + j * 16 + col] = c[vv] + proj_b[oc];
      }
    }
}

// ---------------- launch ---------------------------------------------------

extern "C" void kernel_launch(void* const* d_in, const int* in_sizes, int n_in,
                              void* d_out, int out_size, void* d_ws, size_t ws_size,
                              hipStream_t stream) {
  (void)in_sizes; (void)n_in; (void)out_size; (void)ws_size;
  const float* x      = (const float*)d_in[0];
  const float* qkv_w  = (const float*)d_in[1];
  const float* qkv_s  = (const float*)d_in[2];
  const float* qkv_b  = (const float*)d_in[3];
  const float* pe_w   = (const float*)d_in[4];
  const float* pe_s   = (const float*)d_in[5];
  const float* pe_b   = (const float*)d_in[6];
  const float* proj_w = (const float*)d_in[7];
  const float* proj_s = (const float*)d_in[8];
  const float* proj_b = (const float*)d_in[9];

  char* wp = (char*)d_ws;
  auto alloc = [&](size_t bytes) -> void* {
    void* p = (void*)wp;
    wp += (bytes + 255) & ~(size_t)255;
    return p;
  };
  const size_t xbPBytes = (size_t)8 * PAD_IMG_ELEMS * sizeof(bf16_t);  // 17.8 MB
  bf16_t* Wq    = (bf16_t*)alloc((size_t)9 * 768 * 256 * sizeof(bf16_t));
  bf16_t* Wp    = (bf16_t*)alloc((size_t)256 * 256 * sizeof(bf16_t));
  bf16_t* xbP   = (bf16_t*)alloc(xbPBytes);
  float*  q     = (float*) alloc((size_t)8 * 256 * 4096 * sizeof(float));
  bf16_t* kT    = (bf16_t*)alloc((size_t)64 * 4096 * 32 * sizeof(bf16_t));
  float*  v     = (float*) alloc((size_t)8 * 256 * 4096 * sizeof(float));
  float*  aPool = (float*) alloc((size_t)8 * 256 * 64 * sizeof(float));
  bf16_t* aT    = (bf16_t*)alloc((size_t)64 * 64 * 32 * sizeof(bf16_t));
  float*  attnK = (float*) alloc((size_t)4096 * 4096 * sizeof(float));
  float*  attn  = (float*) alloc((size_t)4096 * 32 * sizeof(float));
  // aliases (lifetimes do not overlap)
  float*  attnOut = attnK;         // attnK dead after k_agg
  float*  yf      = q;             // q dead after k_stage2
  bf16_t* yb      = (bf16_t*)xbP;  // padded image dead after k_qkv_conv

  unsigned n16 = (unsigned)(xbPBytes / 16);
  k_zero16        <<<(n16 + 255) / 256, 256, 0, stream>>>((float4*)xbP, n16);
  k_prep_qkv_w    <<<768,   256, 0, stream>>>(qkv_w, qkv_s, Wq);
  k_prep_proj_w   <<<256,   256, 0, stream>>>(proj_w, proj_s, Wp);
  k_x2padded      <<<4096,  256, 0, stream>>>(x, xbP);
  k_qkv_conv      <<<1536,  256, 0, stream>>>(xbP, Wq, qkv_b, q, kT, v);
  k_agent_pool    <<<512,   256, 0, stream>>>(q, aPool, aT);
  k_stage1        <<<1024,  256, 0, stream>>>(aT, kT, attnK);
  k_softmax       <<<4096,  256, 0, stream>>>(attnK);
  k_agg           <<<4096,  256, 0, stream>>>(attnK, v, attn);
  k_stage2        <<<1024,  256, 0, stream>>>(q, aPool, attn, attnOut);
  k_pe_add        <<<32768, 256, 0, stream>>>(v, attnOut, pe_w, pe_s, pe_b, yf);
  k_nchw2nhwc_bf16<<<4096,  256, 0, stream>>>(yf, yb);
  k_proj          <<<512,   256, 0, stream>>>(yb, Wp, proj_b, (float*)d_out);
}
